// AVWDCRNN_6038724018745
// MI455X (gfx1250) — matrix operand order repaired and checked by reading, not compiled, not run on hardware
//
#include <hip/hip_runtime.h>
#include <math.h>

// ---------- problem constants ----------
#define NN   1024   // nodes
#define BB   64     // batch
#define TT   24     // time steps
#define HH   64     // hidden
#define EE   10     // embedding dim
#define DIN0 1      // layer0 input dim
#define IP0  96     // layer0 padded concat dim (65 -> 96, 3 x K32)
#define IP1  128    // layer1 padded concat dim (128 -> 128)
#define OG   128    // gate output (2H)
#define OU   64     // update output (H)

// ---------- types ----------
typedef __bf16 v16bf __attribute__((ext_vector_type(16)));
typedef __bf16 v8bf  __attribute__((ext_vector_type(8)));
typedef float  v8f   __attribute__((ext_vector_type(8)));

__device__ __forceinline__ __bf16 f2bf(float f) {
    union { float f; unsigned int u; } c; c.f = f;
    unsigned int u = c.u;
    u += 0x7fffu + ((u >> 16) & 1u);          // round-to-nearest-even
    union { unsigned short s; __bf16 b; } r;
    r.s = (unsigned short)(u >> 16);
    return r.b;
}

__device__ __forceinline__ v16bf frag16(const __bf16* p0, const __bf16* p1) {
    v8bf lo = *(const v8bf*)p0;
    v8bf hi = *(const v8bf*)p1;
    return __builtin_shufflevector(lo, hi, 0,1,2,3,4,5,6,7,8,9,10,11,12,13,14,15);
}

__device__ __forceinline__ v8f wmma_bf16(v16bf a, v16bf b, v8f c) {
    return __builtin_amdgcn_wmma_f32_16x16x32_bf16(false, a, false, b, (short)0, c, false, false);
}

// ---------- adjacency: A = softmax(relu(E E^T)) rowwise, as bf16 ----------
__global__ __launch_bounds__(256) void adj_kernel(const float* __restrict__ emb,
                                                  __bf16* __restrict__ Abf) {
    __shared__ float red[256];
    const int n = blockIdx.x, tid = threadIdx.x;
    float en[EE];
    #pragma unroll
    for (int d = 0; d < EE; ++d) en[d] = emb[n * EE + d];

    float v[NN / 256];
    float mx = 0.f;
    #pragma unroll
    for (int j = 0; j < NN / 256; ++j) {
        const int m = tid + j * 256;
        float dot = 0.f;
        #pragma unroll
        for (int d = 0; d < EE; ++d) dot += en[d] * emb[m * EE + d];
        v[j] = fmaxf(dot, 0.f);
        mx = fmaxf(mx, v[j]);
    }
    red[tid] = mx; __syncthreads();
    for (int s = 128; s > 0; s >>= 1) {
        if (tid < s) red[tid] = fmaxf(red[tid], red[tid + s]);
        __syncthreads();
    }
    const float rowmax = red[0]; __syncthreads();

    float sm = 0.f;
    #pragma unroll
    for (int j = 0; j < NN / 256; ++j) sm += __expf(v[j] - rowmax);
    red[tid] = sm; __syncthreads();
    for (int s = 128; s > 0; s >>= 1) {
        if (tid < s) red[tid] += red[tid + s];
        __syncthreads();
    }
    const float inv = 1.f / red[0];

    #pragma unroll
    for (int j = 0; j < NN / 256; ++j) {
        const int m = tid + j * 256;
        Abf[(size_t)n * NN + m] = f2bf(__expf(v[j] - rowmax) * inv);
    }
}

// ---------- per-node weights, pre-transposed: dst[n][k][o][i] = sum_d e[n,d] w[d][k][i][o] ----------
__global__ void wcomb_kernel(const float* __restrict__ emb, const float* __restrict__ w,
                             __bf16* __restrict__ dst, int I, int Ipad, int O) {
    const size_t total = (size_t)NN * 2 * O * Ipad;
    for (size_t idx = (size_t)blockIdx.x * blockDim.x + threadIdx.x;
         idx < total; idx += (size_t)gridDim.x * blockDim.x) {
        const int i = (int)(idx % Ipad);
        size_t r = idx / Ipad;
        const int o = (int)(r % O); r /= O;
        const int kc = (int)(r & 1);
        const int n  = (int)(r >> 1);
        float v = 0.f;
        if (i < I) {
            #pragma unroll
            for (int d = 0; d < EE; ++d)
                v += emb[n * EE + d] * w[(((size_t)d * 2 + kc) * I + i) * O + o];
        }
        dst[idx] = f2bf(v);
    }
}

__global__ void bcomb_kernel(const float* __restrict__ emb, const float* __restrict__ bw,
                             float* __restrict__ dst, int O) {
    const int id = blockIdx.x * blockDim.x + threadIdx.x;
    if (id >= NN * O) return;
    const int n = id / O, o = id % O;
    float v = 0.f;
    #pragma unroll
    for (int d = 0; d < EE; ++d) v += emb[n * EE + d] * bw[d * O + o];
    dst[id] = v;
}

// ---------- init h: (layer,B,N,H) -> (N,B,H) ----------
__global__ void inith_kernel(const float* __restrict__ init, float* __restrict__ hb, int layer) {
    const int id = blockIdx.x * blockDim.x + threadIdx.x;
    if (id >= NN * BB * HH) return;
    const int o = id % HH, b = (id / HH) % BB, n = id / (BB * HH);
    hb[id] = init[(((size_t)layer * BB + b) * NN + n) * HH + o];
}

// ---------- h_last: (N,B,H) -> out (2,B,N,H) ----------
__global__ void hlast_kernel(const float* __restrict__ hb, float* __restrict__ out, int layer) {
    const int id = blockIdx.x * blockDim.x + threadIdx.x;
    if (id >= NN * BB * HH) return;
    const int o = id % HH, b = (id / HH) % BB, n = id / (BB * HH);
    out[(((size_t)layer * BB + b) * NN + n) * HH + o] = hb[id];
}

// ---------- build concat [xt, (z*)h] in row (N,B,Ipad) and col (B*Ipad, N) bf16 layouts ----------
__global__ __launch_bounds__(256) void build_concat_kernel(
        const float* __restrict__ xglob,   // x input (B,T,N,Din) or null
        const float* __restrict__ xnb,     // (N,B,Din) buffer or null
        int Din,
        const float* __restrict__ h,       // (N,B,H)
        const float* __restrict__ z,       // (N,B,H) or null
        __bf16* __restrict__ orow, __bf16* __restrict__ ocol,
        int Ipad, int t) {
    const int id = blockIdx.x * blockDim.x + threadIdx.x;
    if (id >= NN * BB) return;
    const int n = id / BB, b = id % BB;
    const int I = Din + HH;
    for (int c = 0; c < Ipad; ++c) {
        float v;
        if (c < Din) {
            v = xglob ? xglob[((size_t)(b * TT + t) * NN + n) * Din + c]
                      : xnb[((size_t)n * BB + b) * Din + c];
        } else if (c < I) {
            const size_t hi = ((size_t)n * BB + b) * HH + (c - Din);
            v = z ? z[hi] * h[hi] : h[hi];
        } else {
            v = 0.f;
        }
        const __bf16 bv = f2bf(v);
        orow[((size_t)n * BB + b) * Ipad + c] = bv;
        ocol[((size_t)b * Ipad + c) * NN + n] = bv;
    }
}

// ---------- graph GEMM: C(1024 x J) = Adj(1024x1024) @ Xcol(J x 1024)^T, all bf16 ----------
// block = 8 waves; wave tile = 32 rows x 64 cols; block tile = 128 x 128.
__global__ __launch_bounds__(256) void graph_gemm_kernel(
        const __bf16* __restrict__ Adj, const __bf16* __restrict__ Xcol,
        __bf16* __restrict__ C, int J) {
    const int tid = threadIdx.x;
    const int wv = tid >> 5, lane = tid & 31;
    const int l15 = lane & 15, lh = lane >> 4;
    const int m0 = blockIdx.x * 128 + (wv & 3) * 32;
    const int j0 = blockIdx.y * 128 + (wv >> 2) * 64;

    v8f acc[2][4] = {};
    const __bf16* a0 = Adj + (size_t)(m0 + l15) * NN + lh * 8;
    const __bf16* a1 = a0 + (size_t)16 * NN;
    const __bf16* bb = Xcol + (size_t)(j0 + l15) * NN + lh * 8;

    for (int k0 = 0; k0 < NN; k0 += 32) {
        const v16bf fa0 = frag16(a0 + k0, a0 + k0 + 16);
        const v16bf fa1 = frag16(a1 + k0, a1 + k0 + 16);
        #pragma unroll
        for (int ci = 0; ci < 4; ++ci) {
            const __bf16* bp = bb + (size_t)ci * 16 * NN + k0;
            const v16bf fb = frag16(bp, bp + 16);
            acc[0][ci] = wmma_bf16(fa0, fb, acc[0][ci]);
            acc[1][ci] = wmma_bf16(fa1, fb, acc[1][ci]);
        }
    }
    #pragma unroll
    for (int ri = 0; ri < 2; ++ri)
        #pragma unroll
        for (int ci = 0; ci < 4; ++ci)
            #pragma unroll
            for (int r = 0; r < 8; ++r) {
                const int row = m0 + ri * 16 + r + lh * 8;
                const int col = j0 + ci * 16 + l15;
                C[(size_t)row * J + col] = f2bf(acc[ri][ci][r]);
            }
}

// ---------- per-node GEMM: out[n](64 x O) = X0[n]@W[n,0] + X1[n]@W[n,1] (+bias, epilogue) ----------
// block = 4 waves (128 thr); wave handles 16 batch rows x all O columns. NT = O/16.
// MODE 0: sigmoid -> split z|r.  MODE 1: GRU update h = r*h + (1-r)*tanh(.).
template <int NT, int MODE>
__global__ __launch_bounds__(128) void node_gemm_kernel(
        const __bf16* __restrict__ X0, const __bf16* __restrict__ X1,
        const __bf16* __restrict__ WT, const float* __restrict__ bias, int Ipad,
        float* __restrict__ zb, float* __restrict__ rb, float* __restrict__ hb,
        float* __restrict__ outseq, int t) {
    constexpr int O = NT * 16;
    const int n = blockIdx.x;
    const int tid = threadIdx.x, wv = tid >> 5, lane = tid & 31;
    const int l15 = lane & 15, lh = lane >> 4;
    const int b0 = wv * 16;

    v8f acc[NT] = {};
    const __bf16* Ws = WT + (size_t)n * 2 * O * Ipad;

    #pragma unroll
    for (int seg = 0; seg < 2; ++seg) {
        const __bf16* Xs = (seg ? X1 : X0) + (size_t)n * BB * Ipad;
        const __bf16* xp = Xs + (size_t)(b0 + l15) * Ipad + lh * 8;
        const __bf16* wp = Ws + (size_t)seg * O * Ipad;
        for (int k0 = 0; k0 < Ipad; k0 += 32) {
            const v16bf fa = frag16(xp + k0, xp + k0 + 16);
            #pragma unroll
            for (int ct = 0; ct < NT; ++ct) {
                const __bf16* bp = wp + (size_t)(ct * 16 + l15) * Ipad + lh * 8 + k0;
                const v16bf fb = frag16(bp, bp + 16);
                acc[ct] = wmma_bf16(fa, fb, acc[ct]);
            }
        }
    }

    #pragma unroll
    for (int ct = 0; ct < NT; ++ct)
        #pragma unroll
        for (int r = 0; r < 8; ++r) {
            const int o = ct * 16 + l15;
            const int b = b0 + r + lh * 8;
            float v = acc[ct][r] + bias[n * O + o];
            if (MODE == 0) {
                v = 1.f / (1.f + __expf(-v));
                const size_t idx = ((size_t)n * BB + b) * HH + (o & (HH - 1));
                if (o < HH) zb[idx] = v; else rb[idx] = v;
            } else {
                const float hc = tanhf(v);
                const size_t idx = ((size_t)n * BB + b) * HH + o;
                const float rr = rb[idx];
                const float hn = rr * hb[idx] + (1.f - rr) * hc;
                hb[idx] = hn;
                if (outseq) outseq[(((size_t)b * TT + t) * NN + n) * HH + o] = hn;
            }
        }
}

// =====================================================================
extern "C" void kernel_launch(void* const* d_in, const int* in_sizes, int n_in,
                              void* d_out, int out_size, void* d_ws, size_t ws_size,
                              hipStream_t stream) {
    const float* x    = (const float*)d_in[0];
    const float* init = (const float*)d_in[1];
    const float* emb  = (const float*)d_in[2];
    const float* gw0  = (const float*)d_in[3];
    const float* gb0  = (const float*)d_in[4];
    const float* uw0  = (const float*)d_in[5];
    const float* ub0  = (const float*)d_in[6];
    const float* gw1  = (const float*)d_in[7];
    const float* gb1  = (const float*)d_in[8];
    const float* uw1  = (const float*)d_in[9];
    const float* ub1  = (const float*)d_in[10];
    float* out = (float*)d_out;
    (void)in_sizes; (void)n_in; (void)out_size; (void)ws_size;

    // ---- carve workspace (256B aligned) ----
    char* wsp = (char*)d_ws;
    size_t off = 0;
    auto carve = [&](size_t bytes) -> char* {
        char* p = wsp + off;
        off += (bytes + 255) & ~(size_t)255;
        return p;
    };
    __bf16* Abf  = (__bf16*)carve((size_t)NN * NN * 2);
    __bf16* WgT0 = (__bf16*)carve((size_t)NN * 2 * OG * IP0 * 2);
    __bf16* WuT0 = (__bf16*)carve((size_t)NN * 2 * OU * IP0 * 2);
    __bf16* WgT1 = (__bf16*)carve((size_t)NN * 2 * OG * IP1 * 2);
    __bf16* WuT1 = (__bf16*)carve((size_t)NN * 2 * OU * IP1 * 2);
    float*  Bg0  = (float*)carve((size_t)NN * OG * 4);
    float*  Bu0  = (float*)carve((size_t)NN * OU * 4);
    float*  Bg1  = (float*)carve((size_t)NN * OG * 4);
    float*  Bu1  = (float*)carve((size_t)NN * OU * 4);
    float*  h0   = (float*)carve((size_t)NN * BB * HH * 4);
    float*  h1   = (float*)carve((size_t)NN * BB * HH * 4);
    float*  zb   = (float*)carve((size_t)NN * BB * HH * 4);
    float*  rb   = (float*)carve((size_t)NN * BB * HH * 4);
    __bf16* inpR = (__bf16*)carve((size_t)NN * BB * IP1 * 2);
    __bf16* inpC = (__bf16*)carve((size_t)NN * BB * IP1 * 2);
    __bf16* AxR  = (__bf16*)carve((size_t)NN * BB * IP1 * 2);

    // ---- one-time setup ----
    adj_kernel<<<NN, 256, 0, stream>>>(emb, Abf);
    wcomb_kernel<<<8192, 256, 0, stream>>>(emb, gw0, WgT0, DIN0 + HH, IP0, OG);
    wcomb_kernel<<<8192, 256, 0, stream>>>(emb, uw0, WuT0, DIN0 + HH, IP0, OU);
    wcomb_kernel<<<8192, 256, 0, stream>>>(emb, gw1, WgT1, 2 * HH, IP1, OG);
    wcomb_kernel<<<8192, 256, 0, stream>>>(emb, uw1, WuT1, 2 * HH, IP1, OU);
    bcomb_kernel<<<(NN * OG + 255) / 256, 256, 0, stream>>>(emb, gb0, Bg0, OG);
    bcomb_kernel<<<(NN * OU + 255) / 256, 256, 0, stream>>>(emb, ub0, Bu0, OU);
    bcomb_kernel<<<(NN * OG + 255) / 256, 256, 0, stream>>>(emb, gb1, Bg1, OG);
    bcomb_kernel<<<(NN * OU + 255) / 256, 256, 0, stream>>>(emb, ub1, Bu1, OU);
    {
        const int nt = NN * BB * HH, nb = (nt + 255) / 256;
        inith_kernel<<<nb, 256, 0, stream>>>(init, h0, 0);
        inith_kernel<<<nb, 256, 0, stream>>>(init, h1, 1);
    }

    const int J0 = BB * IP0, J1 = BB * IP1;
    const dim3 gg0(NN / 128, J0 / 128), gg1(NN / 128, J1 / 128);
    const int cb = (NN * BB + 255) / 256;

    for (int t = 0; t < TT; ++t) {
        // ---------- layer 0 (xt from input x, Din=1, Ipad=96) ----------
        build_concat_kernel<<<cb, 256, 0, stream>>>(x, nullptr, DIN0, h0, nullptr,
                                                    inpR, inpC, IP0, t);
        graph_gemm_kernel<<<gg0, 256, 0, stream>>>(Abf, inpC, AxR, J0);
        node_gemm_kernel<8, 0><<<NN, 128, 0, stream>>>(inpR, AxR, WgT0, Bg0, IP0,
                                                       zb, rb, h0, nullptr, t);
        build_concat_kernel<<<cb, 256, 0, stream>>>(x, nullptr, DIN0, h0, zb,
                                                    inpR, inpC, IP0, t);
        graph_gemm_kernel<<<gg0, 256, 0, stream>>>(Abf, inpC, AxR, J0);
        node_gemm_kernel<4, 1><<<NN, 128, 0, stream>>>(inpR, AxR, WuT0, Bu0, IP0,
                                                       zb, rb, h0, nullptr, t);
        // ---------- layer 1 (xt = layer0 output h0, Din=64, Ipad=128) ----------
        build_concat_kernel<<<cb, 256, 0, stream>>>(nullptr, h0, HH, h1, nullptr,
                                                    inpR, inpC, IP1, t);
        graph_gemm_kernel<<<gg1, 256, 0, stream>>>(Abf, inpC, AxR, J1);
        node_gemm_kernel<8, 0><<<NN, 128, 0, stream>>>(inpR, AxR, WgT1, Bg1, IP1,
                                                       zb, rb, h1, nullptr, t);
        build_concat_kernel<<<cb, 256, 0, stream>>>(nullptr, h0, HH, h1, zb,
                                                    inpR, inpC, IP1, t);
        graph_gemm_kernel<<<gg1, 256, 0, stream>>>(Abf, inpC, AxR, J1);
        node_gemm_kernel<4, 1><<<NN, 128, 0, stream>>>(inpR, AxR, WuT1, Bu1, IP1,
                                                       zb, rb, h1, out /*out_seq*/, t);
    }

    // ---- h_last -> out tail (2,B,N,H) ----
    float* hid = out + (size_t)BB * TT * NN * HH;
    const int nt = NN * BB * HH, nb = (nt + 255) / 256;
    hlast_kernel<<<nb, 256, 0, stream>>>(h0, hid, 0);
    hlast_kernel<<<nb, 256, 0, stream>>>(h1, hid, 1);
}
